// Seq2Seq_56341380989528
// MI455X (gfx1250) — compile-verified, matrix operand order checked by reading, not run on hardware
//
#include <hip/hip_runtime.h>

// Problem constants (match reference)
#define B_   64
#define T_   512
#define D_   128
#define H_   512
#define C_   256

typedef __attribute__((ext_vector_type(16))) __bf16       v16bf;
typedef __attribute__((ext_vector_type(8)))  float        v8f;
typedef __attribute__((ext_vector_type(4)))  unsigned int u32x4;

union ABu { v16bf v; u32x4 q[2]; };

__device__ __forceinline__ unsigned short f2bf(float f) {
  unsigned int u = __float_as_uint(f);
  u += 0x7FFFu + ((u >> 16) & 1u);   // round-to-nearest-even
  return (unsigned short)(u >> 16);
}

__device__ __forceinline__ float sigmoidf_(float x) {
  return 1.0f / (1.0f + __expf(-x));
}

__device__ __forceinline__ v8f splat8(float x) {
  v8f r = {x, x, x, x, x, x, x, x};
  return r;
}

// A operand (16x32 bf16, rows m=lane&15): per CDNA5 ISA 16-bit A layout,
// lanes 0-15 hold K 0..7 / 16..23 (v0-3 / v4-7), lanes 16-31 hold K 8..15 / 24..31.
// Row-major source => two contiguous 16B loads per lane.
__device__ __forceinline__ v16bf loadA(const unsigned short* base, int ld,
                                       int row0, int k0, int m, int hi) {
  const unsigned short* p = base + (row0 + m) * ld + k0 + 8 * hi;
  ABu u;
  u.q[0] = *(const u32x4*)(p);
  u.q[1] = *(const u32x4*)(p + 16);
  return u.v;
}

// B operand: pre-packed so each lane loads 16 consecutive bf16 (32B) per tile.
__device__ __forceinline__ v16bf loadB(const unsigned short* tilebase, int lane) {
  const unsigned short* p = tilebase + lane * 16;
  ABu u;
  u.q[0] = *(const u32x4*)(p);
  u.q[1] = *(const u32x4*)(p + 8);
  return u.v;
}

// ---------------------------------------------------------------------------
// Pack W (row-major [N x K] fp32) into WMMA-B tile layout, bf16.
// ---------------------------------------------------------------------------
__global__ void pack_b_kernel(const float* __restrict__ src,
                              unsigned short* __restrict__ dst, int N, int K) {
  int idx = blockIdx.x * blockDim.x + threadIdx.x;
  if (idx >= N * K) return;
  int tile = idx >> 9;
  int r    = idx & 511;
  int l    = r >> 4;
  int e    = r & 15;
  int v    = e >> 1;
  int hf   = e & 1;
  int ktiles = K >> 5;
  int kt = tile % ktiles;
  int nt = tile / ktiles;
  int n = nt * 16 + (l & 15);
  int k = kt * 32 + ((v >= 4) ? 16 : 0) + ((v & 3) * 2) + hf + 8 * (l >> 4);
  dst[idx] = f2bf(src[(size_t)n * K + k]);
}

// outputs[:, 0, :] = 0
__global__ void zero_t0_kernel(float* __restrict__ out) {
  int i = blockIdx.x * blockDim.x + threadIdx.x;
  if (i < B_ * D_) {
    int b = i >> 7, d = i & 127;
    out[(size_t)b * T_ * D_ + d] = 0.0f;
  }
}

// ---------------------------------------------------------------------------
// One LSTM layer step.  Per group: accumulate the 4 gate tiles (only 32 acc
// VGPRs live), run the cell update immediately, park the new h tile as 4
// packed dwords (8 bf16).  After all groups: barrier, then scatter the packed
// h values to LDS (and dec_out), barrier.
//
// A laundered zero *offset* (not a laundered pointer!) is added to the weight
// bases each call: the loads stay rooted at the kernarg pointer so they lower
// to global_load_b128 (LOADcnt only, no flat/DScnt coupling), but the offset
// is loop-variant so LICM cannot hoist the time-invariant weight tiles out of
// the sequence loop and spill them to scratch.
// ---------------------------------------------------------------------------
__device__ __forceinline__ void lstm_layer(
    const unsigned short* __restrict__ Ax, int ldx, int kxt,
    const unsigned short* __restrict__ Wx,
    const unsigned short* __restrict__ Ah,
    const unsigned short* __restrict__ Wh,
    const float* __restrict__ bias,
    float cst[4][8],
    unsigned short* __restrict__ hdst,
    unsigned short* __restrict__ decout, int tstep,
    int w, int lane) {
  int m = lane & 15, hi = lane >> 4;
  unsigned int hpk[4][4];  // packed new-h: [group][r/2], 2 bf16 per dword

  size_t zx = 0, zh = 0;
  asm volatile("" : "+s"(zx), "+s"(zh));
  const unsigned short* WxL = Wx + zx;
  const unsigned short* WhL = Wh + zh;

#pragma unroll
  for (int gi = 0; gi < 4; ++gi) {
    int group = w * 4 + gi;
    int rt = group >> 5;
    int jt = group & 31;

    v8f accI = splat8(bias[0 * H_ + jt * 16 + m]);
    v8f accF = splat8(bias[1 * H_ + jt * 16 + m]);
    v8f accG = splat8(bias[2 * H_ + jt * 16 + m]);
    v8f accO = splat8(bias[3 * H_ + jt * 16 + m]);

    // input path (x for layer0, h0 for layer1)
    for (int kt = 0; kt < kxt; ++kt) {
      v16bf a = loadA(Ax, ldx, rt * 16, kt * 32, m, hi);
      const unsigned short* wb = WxL + ((size_t)(jt * kxt + kt) << 9);
      size_t gstride = (size_t)(32 * kxt) << 9;
      accI = __builtin_amdgcn_wmma_f32_16x16x32_bf16(
          false, a, false, loadB(wb, lane), (short)0, accI, false, false);
      accF = __builtin_amdgcn_wmma_f32_16x16x32_bf16(
          false, a, false, loadB(wb + gstride, lane), (short)0, accF, false, false);
      accG = __builtin_amdgcn_wmma_f32_16x16x32_bf16(
          false, a, false, loadB(wb + 2 * gstride, lane), (short)0, accG, false, false);
      accO = __builtin_amdgcn_wmma_f32_16x16x32_bf16(
          false, a, false, loadB(wb + 3 * gstride, lane), (short)0, accO, false, false);
    }
    // recurrent path (K = 512 -> 16 tiles)
    for (int kt = 0; kt < 16; ++kt) {
      v16bf a = loadA(Ah, H_, rt * 16, kt * 32, m, hi);
      const unsigned short* wb = WhL + ((size_t)(jt * 16 + kt) << 9);
      const size_t gstride = (size_t)(32 * 16) << 9;
      accI = __builtin_amdgcn_wmma_f32_16x16x32_bf16(
          false, a, false, loadB(wb, lane), (short)0, accI, false, false);
      accF = __builtin_amdgcn_wmma_f32_16x16x32_bf16(
          false, a, false, loadB(wb + gstride, lane), (short)0, accF, false, false);
      accG = __builtin_amdgcn_wmma_f32_16x16x32_bf16(
          false, a, false, loadB(wb + 2 * gstride, lane), (short)0, accG, false, false);
      accO = __builtin_amdgcn_wmma_f32_16x16x32_bf16(
          false, a, false, loadB(wb + 3 * gstride, lane), (short)0, accO, false, false);
    }

    // cell update, immediately after this group's accumulation
#pragma unroll
    for (int rr = 0; rr < 4; ++rr) {
      unsigned int pk = 0;
#pragma unroll
      for (int p = 0; p < 2; ++p) {
        int r = rr * 2 + p;
        float iv = sigmoidf_(accI[r]);
        float fv = sigmoidf_(accF[r]);
        float gv = tanhf(accG[r]);
        float ov = sigmoidf_(accO[r]);
        float c  = fv * cst[gi][r] + iv * gv;
        cst[gi][r] = c;
        float h = ov * tanhf(c);
        pk |= (unsigned int)f2bf(h) << (16 * p);
      }
      hpk[gi][rr] = pk;
    }
  }

  __syncthreads();  // all waves finished reading Ax/Ah

#pragma unroll
  for (int gi = 0; gi < 4; ++gi) {
    int group = w * 4 + gi;
    int rt = group >> 5;
    int jt = group & 31;
    int col = jt * 16 + m;
#pragma unroll
    for (int rr = 0; rr < 4; ++rr) {
      unsigned int pk = hpk[gi][rr];
#pragma unroll
      for (int p = 0; p < 2; ++p) {
        int r = rr * 2 + p;
        int row = rt * 16 + r + 8 * hi;
        unsigned short hb = (unsigned short)(pk >> (16 * p));
        hdst[row * H_ + col] = hb;
        if (decout) decout[((size_t)tstep * B_ + row) * H_ + col] = hb;
      }
    }
  }
  __syncthreads();  // new h visible
}

// ---------------------------------------------------------------------------
// Persistent recurrent kernel: single 1024-thread workgroup (32 waves) runs
// encoder (512 steps) then decoder (511 steps); h0/h1 live in LDS as bf16,
// c0/c1 live in VGPRs. Decoder layer-1 h is streamed to dec_out (bf16).
// LDS: 2*64KB (h) + 16KB (x tile) = 144KB of the WGP budget.
// ---------------------------------------------------------------------------
__global__ __launch_bounds__(1024, 1) void lstm_seq_kernel(
    const float* __restrict__ x, const float* __restrict__ y,
    const unsigned short* __restrict__ eWih0, const unsigned short* __restrict__ eWhh0,
    const float* __restrict__ eb0,
    const unsigned short* __restrict__ eWih1, const unsigned short* __restrict__ eWhh1,
    const float* __restrict__ eb1,
    const unsigned short* __restrict__ dWih0, const unsigned short* __restrict__ dWhh0,
    const float* __restrict__ db0,
    const unsigned short* __restrict__ dWih1, const unsigned short* __restrict__ dWhh1,
    const float* __restrict__ db1,
    unsigned short* __restrict__ dec_out) {
  __shared__ __align__(16) unsigned short h0s[B_ * H_];   // 64 KB
  __shared__ __align__(16) unsigned short h1s[B_ * H_];   // 64 KB
  __shared__ __align__(16) unsigned short xss[B_ * D_];   // 16 KB

  int tid  = threadIdx.x;
  int lane = tid & 31;
  int w    = tid >> 5;

  for (int i = tid; i < B_ * H_; i += 1024) {
    h0s[i] = 0;
    h1s[i] = 0;
  }
  float c0[4][8], c1[4][8];
#pragma unroll
  for (int gi = 0; gi < 4; ++gi)
#pragma unroll
    for (int r = 0; r < 8; ++r) {
      c0[gi][r] = 0.0f;
      c1[gi][r] = 0.0f;
    }
  __syncthreads();

  for (int phase = 0; phase < 2; ++phase) {
    const float* xin = phase ? y : x;
    int TT = phase ? (T_ - 1) : T_;
    const unsigned short* Wih0 = phase ? dWih0 : eWih0;
    const unsigned short* Whh0 = phase ? dWhh0 : eWhh0;
    const unsigned short* Wih1 = phase ? dWih1 : eWih1;
    const unsigned short* Whh1 = phase ? dWhh1 : eWhh1;
    const float* b0 = phase ? db0 : eb0;
    const float* b1 = phase ? db1 : eb1;

    for (int t = 0; t < TT; ++t) {
      // stage x_t (x layout [B, T, D])
      for (int i = tid; i < B_ * D_; i += 1024) {
        int b = i >> 7, d = i & 127;
        xss[i] = f2bf(xin[((size_t)b * T_ + t) * D_ + d]);
      }
      __syncthreads();

      lstm_layer(xss, D_, D_ / 32, Wih0, h0s, Whh0, b0, c0, h0s,
                 nullptr, 0, w, lane);
      lstm_layer(h0s, H_, H_ / 32, Wih1, h1s, Whh1, b1, c1, h1s,
                 phase ? dec_out : nullptr, t, w, lane);
    }
  }
}

// ---------------------------------------------------------------------------
// Fused classifier: per 16-row tile of dec_out (rows = t*64+b, 2044 tiles):
//   hid = relu(dec @ W1^T + b1)   [16 x 256]  (K=512)
//   out = hid @ W2^T + b2         [16 x 128]  (K=256)
// 256 threads = 8 waves; hid kept as bf16 in LDS between the two GEMMs.
// ---------------------------------------------------------------------------
__global__ __launch_bounds__(256, 2) void clf_kernel(
    const unsigned short* __restrict__ dec,
    const unsigned short* __restrict__ pW1, const float* __restrict__ b1,
    const unsigned short* __restrict__ pW2, const float* __restrict__ b2,
    float* __restrict__ out) {
  __shared__ __align__(16) unsigned short As[16 * H_];  // 16 KB
  __shared__ __align__(16) unsigned short Hs[16 * C_];  // 8 KB

  int tid  = threadIdx.x;
  int lane = tid & 31;
  int w    = tid >> 5;
  int m    = lane & 15;
  int hi   = lane >> 4;
  size_t row0 = (size_t)blockIdx.x * 16;

  // stage 16 rows x 512 bf16 of dec_out
  {
    const u32x4* src = (const u32x4*)(dec + row0 * H_);
    u32x4* dst = (u32x4*)As;
    for (int i = tid; i < (16 * H_) / 8; i += 256) dst[i] = src[i];
  }
  __syncthreads();

  // hidden layer: 16 col-tiles, 2 per wave
#pragma unroll
  for (int cc = 0; cc < 2; ++cc) {
    int ct = w * 2 + cc;
    v8f acc = splat8(b1[ct * 16 + m]);
    for (int kt = 0; kt < H_ / 32; ++kt) {
      v16bf a = loadA(As, H_, 0, kt * 32, m, hi);
      v16bf b = loadB(pW1 + ((size_t)(ct * (H_ / 32) + kt) << 9), lane);
      acc = __builtin_amdgcn_wmma_f32_16x16x32_bf16(false, a, false, b,
                                                    (short)0, acc, false, false);
    }
#pragma unroll
    for (int r = 0; r < 8; ++r) {
      float v = acc[r];
      v = v > 0.0f ? v : 0.0f;  // ReLU
      Hs[(r + 8 * hi) * C_ + ct * 16 + m] = f2bf(v);
    }
  }
  __syncthreads();

  // output layer: 8 col-tiles, 1 per wave
  {
    v8f acc = splat8(b2[w * 16 + m]);
    for (int kt = 0; kt < C_ / 32; ++kt) {
      v16bf a = loadA(Hs, C_, 0, kt * 32, m, hi);
      v16bf b = loadB(pW2 + ((size_t)(w * (C_ / 32) + kt) << 9), lane);
      acc = __builtin_amdgcn_wmma_f32_16x16x32_bf16(false, a, false, b,
                                                    (short)0, acc, false, false);
    }
#pragma unroll
    for (int r = 0; r < 8; ++r) {
      int row = (int)row0 + r + 8 * hi;  // row = t*64 + b
      int t = row >> 6;
      int b = row & 63;
      out[((size_t)b * T_ + (t + 1)) * D_ + w * 16 + m] = acc[r];
    }
  }
}

// ---------------------------------------------------------------------------
extern "C" void kernel_launch(void* const* d_in, const int* in_sizes, int n_in,
                              void* d_out, int out_size, void* d_ws, size_t ws_size,
                              hipStream_t stream) {
  (void)in_sizes; (void)n_in; (void)out_size; (void)ws_size;

  const float* x        = (const float*)d_in[0];
  const float* y        = (const float*)d_in[1];
  const float* enc_Wih0 = (const float*)d_in[2];
  const float* enc_Whh0 = (const float*)d_in[3];
  const float* enc_b0   = (const float*)d_in[4];
  const float* enc_Wih1 = (const float*)d_in[5];
  const float* enc_Whh1 = (const float*)d_in[6];
  const float* enc_b1   = (const float*)d_in[7];
  const float* dec_Wih0 = (const float*)d_in[8];
  const float* dec_Whh0 = (const float*)d_in[9];
  const float* dec_b0   = (const float*)d_in[10];
  const float* dec_Wih1 = (const float*)d_in[11];
  const float* dec_Whh1 = (const float*)d_in[12];
  const float* dec_b1   = (const float*)d_in[13];
  const float* clf_W1   = (const float*)d_in[14];
  const float* clf_b1   = (const float*)d_in[15];
  const float* clf_W2   = (const float*)d_in[16];
  const float* clf_b2   = (const float*)d_in[17];
  float* out = (float*)d_out;

  // workspace layout (bf16 packed weights + decoder output), ~45.3 MB total
  char* ws = (char*)d_ws;
  size_t off = 0;
  auto take = [&](size_t bytes) {
    char* p = ws + off;
    off += (bytes + 255) & ~(size_t)255;
    return p;
  };
  unsigned short* p_eWih0 = (unsigned short*)take((size_t)4 * H_ * D_ * 2);
  unsigned short* p_eWhh0 = (unsigned short*)take((size_t)4 * H_ * H_ * 2);
  unsigned short* p_eWih1 = (unsigned short*)take((size_t)4 * H_ * H_ * 2);
  unsigned short* p_eWhh1 = (unsigned short*)take((size_t)4 * H_ * H_ * 2);
  unsigned short* p_dWih0 = (unsigned short*)take((size_t)4 * H_ * D_ * 2);
  unsigned short* p_dWhh0 = (unsigned short*)take((size_t)4 * H_ * H_ * 2);
  unsigned short* p_dWih1 = (unsigned short*)take((size_t)4 * H_ * H_ * 2);
  unsigned short* p_dWhh1 = (unsigned short*)take((size_t)4 * H_ * H_ * 2);
  unsigned short* p_W1    = (unsigned short*)take((size_t)C_ * H_ * 2);
  unsigned short* p_W2    = (unsigned short*)take((size_t)D_ * C_ * 2);
  unsigned short* dec_out = (unsigned short*)take((size_t)(T_ - 1) * B_ * H_ * 2);

  auto pack = [&](const float* src, unsigned short* dst, int N, int K) {
    int n = N * K;
    hipLaunchKernelGGL(pack_b_kernel, dim3((n + 255) / 256), dim3(256), 0,
                       stream, src, dst, N, K);
  };
  pack(enc_Wih0, p_eWih0, 4 * H_, D_);
  pack(enc_Whh0, p_eWhh0, 4 * H_, H_);
  pack(enc_Wih1, p_eWih1, 4 * H_, H_);
  pack(enc_Whh1, p_eWhh1, 4 * H_, H_);
  pack(dec_Wih0, p_dWih0, 4 * H_, D_);
  pack(dec_Whh0, p_dWhh0, 4 * H_, H_);
  pack(dec_Wih1, p_dWih1, 4 * H_, H_);
  pack(dec_Whh1, p_dWhh1, 4 * H_, H_);
  pack(clf_W1, p_W1, C_, H_);
  pack(clf_W2, p_W2, D_, C_);

  hipLaunchKernelGGL(zero_t0_kernel, dim3((B_ * D_ + 255) / 256), dim3(256), 0,
                     stream, out);

  hipLaunchKernelGGL(lstm_seq_kernel, dim3(1), dim3(1024), 0, stream,
                     x, y,
                     p_eWih0, p_eWhh0, enc_b0, p_eWih1, p_eWhh1, enc_b1,
                     p_dWih0, p_dWhh0, dec_b0, p_dWih1, p_dWhh1, dec_b1,
                     dec_out);

  int row_tiles = (B_ * (T_ - 1)) / 16;  // 2044
  hipLaunchKernelGGL(clf_kernel, dim3(row_tiles), dim3(256), 0, stream,
                     dec_out, p_W1, clf_b1, p_W2, clf_b2, out);
}
